// StyleLossPass1_71064528879744
// MI455X (gfx1250) — matrix operand order repaired and checked by reading, not compiled, not run on hardware
//
#include <hip/hip_runtime.h>
#include <hip/hip_bf16.h>

// ---------- types ----------
typedef __attribute__((ext_vector_type(16))) __bf16 v16bf;
typedef __attribute__((ext_vector_type(8)))  __bf16 v8bf;
typedef __attribute__((ext_vector_type(8)))  float  v8f;
typedef __attribute__((ext_vector_type(2)))  float  v2f;

// ---------- problem constants ----------
#define P_    7056      // img patches (84*84)
#define Q_    7056      // style patches
#define D_    4608      // C * 9
#define NP_   84
#define C_    512
#define W_    252
#define HW_   63504     // 252*252
#define HWP_  63552     // padded to multiple of 32 (48 zero cols)
#define PTILE_ 128
#define QTILE_ 128
#define NSEG_  8
#define QT_PER_SEG_ 7   // 56 q-tiles / 8 segments
#define KC2_   64       // scores K chunk (2 bf16 wmma steps)
#define NK2_   (D_ / KC2_)      // 72
#define SROW_  72       // padded LDS row stride (ushorts): 144B -> conflict-free
#define SBUF_  (128 * SROW_)
#define GKC_   32       // gram K chunk (floats)
#define GROW_  36       // padded LDS row stride (floats): 144B
#define GBUF_  (128 * GROW_)
#define GKCH_  1986     // HWP_/32
#define NKSEG_ 16
#define KSEG_LEN_ 125   // ceil(1986/16)

// ---------- workspace layout (bytes) ----------
static constexpr size_t SZ_BF    = (size_t)P_ * D_ * 2;       // 65,028,096 per tensor
static constexpr size_t OFF_IMG  = SZ_BF;
static constexpr size_t SZ_M     = (size_t)C_ * HWP_ * 4;     // 130,154,496 (aliases bf16 bufs)
static constexpr size_t OFF_IDX  = SZ_M;
static constexpr size_t OFF_MSUM = OFF_IDX + 32768;
static constexpr size_t OFF_CV   = OFF_MSUM + 256;
static constexpr size_t OFF_CI   = OFF_CV + (size_t)7168 * NSEG_ * 4;
static constexpr size_t OFF_PART = OFF_CI + (size_t)7168 * NSEG_ * 4;
// total ≈ 147.4 MB

static __device__ __forceinline__ unsigned short f2bf(float f) {
  unsigned int u = __float_as_uint(f);
  u += 0x7FFF + ((u >> 16) & 1);              // round-to-nearest-even
  return (unsigned short)(u >> 16);
}

// 64B async Global->LDS copy: 4x GLOBAL_LOAD_ASYNC_TO_LDS_B128.
// INST_OFFSET applies to BOTH the LDS and global address (ISA 08_async_tensor §4.4),
// so one base pair covers all four beats. Tracked by ASYNCcnt.
static __device__ __forceinline__ void async_ld64(unsigned lds, unsigned long long g) {
  asm volatile(
      "global_load_async_to_lds_b128 %0, %1, off\n\t"
      "global_load_async_to_lds_b128 %0, %1, off offset:16\n\t"
      "global_load_async_to_lds_b128 %0, %1, off offset:32\n\t"
      "global_load_async_to_lds_b128 %0, %1, off offset:48"
      :: "v"(lds), "v"(g) : "memory");
}
static __device__ __forceinline__ void wait_async0() {
  asm volatile("s_wait_asynccnt 0" ::: "memory");
}
static __device__ __forceinline__ unsigned lds_off(const void* p) {
  // generic LDS pointer: low 32 bits are the allocation-relative LDS byte offset
  return (unsigned)(unsigned long long)p;
}

// ============ kernel 1: f32 feature maps -> bf16 patch-major [P][D] ============
__global__ __launch_bounds__(256) void pack_patches_k(
    const float* __restrict__ style, const float* __restrict__ img,
    unsigned short* __restrict__ styb, unsigned short* __restrict__ imgb) {
  const int total = P_ * D_;
  for (int i = blockIdx.x * blockDim.x + threadIdx.x; i < total;
       i += gridDim.x * blockDim.x) {
    int p  = i / D_;
    int k  = i - p * D_;
    int c  = k / 9;
    int rm = k - c * 9;
    int dy = rm / 3, dx = rm - dy * 3;
    int ph = p / NP_, pw = p - ph * NP_;
    int src = c * HW_ + (ph * 3 + dy) * W_ + (pw * 3 + dx);
    styb[i] = f2bf(style[src]);
    imgb[i] = f2bf(img[src]);
  }
}

// ============ kernel 2: bf16 WMMA scores GEMM + fused per-row argmax ============
// grid: (56 p-tiles, 8 q-segments), block: 256 threads = 8 waves (4x2 wave grid).
// Async double-buffered LDS: issue K-chunk kt+1 while WMMA-ing chunk kt.
__global__ __launch_bounds__(256) void scores_argmax_k(
    const unsigned short* __restrict__ A,  // img_bf16 [P][D]
    const unsigned short* __restrict__ B,  // sty_bf16 [Q][D]
    float* __restrict__ cval, int* __restrict__ cidx) {
  __shared__ __align__(16) unsigned short As[2 * SBUF_];  // 36 KB
  __shared__ __align__(16) unsigned short Bs[2 * SBUF_];  // 36 KB
  __shared__ float s_mv[256];
  __shared__ int   s_mi[256];

  const int tid  = threadIdx.x;
  const int lane = tid & 31, wave = tid >> 5;
  const int wm = wave >> 1, wn = wave & 1;
  const int lm = lane & 15, hi = lane >> 4;
  const int row0 = blockIdx.x * PTILE_;
  const int seg  = blockIdx.y;

  const int ldr  = tid >> 1;           // 0..127 : tile row this thread stages
  const int half = (tid & 1) * 32;     // 0/32 ushorts : 64B half-row

  // clamp: garbage rows/cols are filtered at argmax (q<Q) and at the final write (p<P)
  const int gp = (row0 + ldr < P_) ? (row0 + ldr) : (P_ - 1);
  const unsigned lda0 = lds_off(&As[0 * SBUF_ + ldr * SROW_ + half]);
  const unsigned lda1 = lds_off(&As[1 * SBUF_ + ldr * SROW_ + half]);
  const unsigned ldb0 = lds_off(&Bs[0 * SBUF_ + ldr * SROW_ + half]);
  const unsigned ldb1 = lds_off(&Bs[1 * SBUF_ + ldr * SROW_ + half]);

  float best_v = -__builtin_inff();
  int   best_i = 0;

  const __bf16* As_bf = reinterpret_cast<const __bf16*>(As);
  const __bf16* Bs_bf = reinterpret_cast<const __bf16*>(Bs);

  for (int qt = 0; qt < QT_PER_SEG_; ++qt) {
    const int qbase = (seg * QT_PER_SEG_ + qt) * QTILE_;
    const int gq = (qbase + ldr < Q_) ? (qbase + ldr) : (Q_ - 1);
    const unsigned long long gA = (unsigned long long)(A + (size_t)gp * D_ + half);
    const unsigned long long gB = (unsigned long long)(B + (size_t)gq * D_ + half);

    v8f acc[2][4];
    const v8f vzero = {0.f, 0.f, 0.f, 0.f, 0.f, 0.f, 0.f, 0.f};
#pragma unroll
    for (int tm = 0; tm < 2; ++tm)
#pragma unroll
      for (int tn = 0; tn < 4; ++tn) acc[tm][tn] = vzero;

    // prologue: stage chunk 0 into buffer 0
    async_ld64(lda0, gA);
    async_ld64(ldb0, gB);

    int cur = 0;
    for (int kt = 0; kt < NK2_; ++kt) {
      wait_async0();        // my current-buffer loads complete
      __syncthreads();      // everyone's loads done + everyone done reading other buf
      if (kt + 1 < NK2_) {  // prefetch next chunk into the other buffer
        unsigned long long off = (unsigned long long)((kt + 1) * KC2_ * 2);
        async_ld64(cur ? lda0 : lda1, gA + off);
        async_ld64(cur ? ldb0 : ldb1, gB + off);
      }
      // ---- 2 wmma K-steps on current buffer ----
#pragma unroll
      for (int ks = 0; ks < 2; ++ks) {
        const int ko = ks * 32;
        v16bf bmat[4];
#pragma unroll
        for (int tn = 0; tn < 4; ++tn) {
          const __bf16* bp =
              Bs_bf + cur * SBUF_ + (wn * 64 + tn * 16 + lm) * SROW_ + ko + hi * 8;
          v8bf b0 = *reinterpret_cast<const v8bf*>(bp);
          v8bf b1 = *reinterpret_cast<const v8bf*>(bp + 16);
          bmat[tn] = __builtin_shufflevector(b0, b1, 0, 1, 2, 3, 4, 5, 6, 7, 8, 9,
                                             10, 11, 12, 13, 14, 15);
        }
#pragma unroll
        for (int tm = 0; tm < 2; ++tm) {
          const __bf16* ap =
              As_bf + cur * SBUF_ + (wm * 32 + tm * 16 + lm) * SROW_ + ko + hi * 8;
          v8bf a0 = *reinterpret_cast<const v8bf*>(ap);
          v8bf a1 = *reinterpret_cast<const v8bf*>(ap + 16);
          v16bf amat = __builtin_shufflevector(a0, a1, 0, 1, 2, 3, 4, 5, 6, 7, 8, 9,
                                               10, 11, 12, 13, 14, 15);
#pragma unroll
          for (int tn = 0; tn < 4; ++tn) {
            acc[tm][tn] = __builtin_amdgcn_wmma_f32_16x16x32_bf16(
                false, amat, false, bmat[tn], (short)0, acc[tm][tn], false, false);
          }
        }
      }
      cur ^= 1;
    }

    // ---- fused argmax: in-lane over tn, then xor-butterfly over 16-lane half ----
#pragma unroll
    for (int tm = 0; tm < 2; ++tm) {
#pragma unroll
      for (int r = 0; r < 8; ++r) {
        float v = -__builtin_inff();
        int   qi = 0;
#pragma unroll
        for (int tn = 0; tn < 4; ++tn) {
          int q = qbase + wn * 64 + tn * 16 + lm;
          float x = acc[tm][tn][r];
          if (q < Q_ && (x > v || (x == v && q < qi))) { v = x; qi = q; }
        }
#pragma unroll
        for (int off = 8; off >= 1; off >>= 1) {
          float ov = __shfl_xor(v, off, 32);
          int   oi = __shfl_xor(qi, off, 32);
          if (ov > v || (ov == v && oi < qi)) { v = ov; qi = oi; }
        }
        if (lm == 0) {
          int rowl = wm * 32 + tm * 16 + hi * 8 + r;  // C-layout: VGPR r -> M=r / r+8
          s_mv[rowl * 2 + wn] = v;
          s_mi[rowl * 2 + wn] = qi;
        }
      }
    }
    __syncthreads();
    if (tid < 128) {
#pragma unroll
      for (int j = 0; j < 2; ++j) {
        float v = s_mv[tid * 2 + j];
        int   qi = s_mi[tid * 2 + j];
        if (v > best_v || (v == best_v && qi < best_i)) { best_v = v; best_i = qi; }
      }
    }
    __syncthreads();
  }

  if (tid < 128) {
    int p = row0 + tid;
    if (p < P_) {
      cval[(size_t)p * NSEG_ + seg] = best_v;
      cidx[(size_t)p * NSEG_ + seg] = best_i;
    }
  }
}

// ============ kernel 3: merge segment candidates -> idx[P] ============
__global__ __launch_bounds__(256) void merge_argmax_k(
    const float* __restrict__ cval, const int* __restrict__ cidx,
    int* __restrict__ idx) {
  int p = blockIdx.x * blockDim.x + threadIdx.x;
  if (p >= P_) return;
  float bv = -__builtin_inff();
  int   bi = 0;
#pragma unroll
  for (int s = 0; s < NSEG_; ++s) {
    float v = cval[(size_t)p * NSEG_ + s];
    int   i = cidx[(size_t)p * NSEG_ + s];
    if (v > bv || (v == bv && i < bi)) { bv = v; bi = i; }
  }
  idx[p] = bi;
}

// ============ kernel 4: gather matched patches * mask -> M[C][HWP] f32 ============
__global__ __launch_bounds__(256) void gather_mask_k(
    const float* __restrict__ style, const float* __restrict__ mask,
    const int* __restrict__ idx, float* __restrict__ M) {
  const int total = C_ * HWP_;
  for (int i = blockIdx.x * blockDim.x + threadIdx.x; i < total;
       i += gridDim.x * blockDim.x) {
    int c  = i / HWP_;
    int hw = i - c * HWP_;
    float val = 0.f;
    if (hw < HW_) {
      int h = hw / W_, w = hw - h * W_;
      int ph = h / 3, pw = w / 3;
      int q  = idx[ph * NP_ + pw];
      int qh = q / NP_, qw = q - qh * NP_;
      int sh = qh * 3 + (h - ph * 3);
      int sw = qw * 3 + (w - pw * 3);
      val = style[c * HW_ + sh * W_ + sw] * mask[hw];
    }
    M[i] = val;
  }
}

// ============ kernel 5: sum(mask) (single block, deterministic) ============
__global__ __launch_bounds__(256) void mask_sum_k(const float* __restrict__ mask,
                                                  float* __restrict__ out) {
  __shared__ float red[256];
  float s = 0.f;
  for (int i = threadIdx.x; i < HW_; i += 256) s += mask[i];
  red[threadIdx.x] = s;
  __syncthreads();
  for (int o = 128; o > 0; o >>= 1) {
    if (threadIdx.x < o) red[threadIdx.x] += red[threadIdx.x + o];
    __syncthreads();
  }
  if (threadIdx.x == 0) out[0] = red[0];
}

// ============ kernel 6: gram split-K partials via f32 WMMA (async double-buffer) ====
// grid: (16 output tiles of 128x128, 16 K-segments). block: 256 = 8 waves (4x2).
__global__ __launch_bounds__(256) void gram_partial_k(
    const float* __restrict__ M, float* __restrict__ part) {
  __shared__ __align__(16) float As[2 * GBUF_];  // 36 KB
  __shared__ __align__(16) float Bs[2 * GBUF_];  // 36 KB

  const int tid  = threadIdx.x;
  const int lane = tid & 31, wave = tid >> 5;
  const int wm = wave >> 1, wn = wave & 1;
  const int lm = lane & 15, hi = lane >> 4;
  const int tile = blockIdx.x;
  const int ti = tile >> 2, tj = tile & 3;
  const int c0 = ti * 128, d0 = tj * 128;
  const int kseg = blockIdx.y;
  const int kbeg = kseg * KSEG_LEN_;
  const int kend = (kbeg + KSEG_LEN_ < GKCH_) ? (kbeg + KSEG_LEN_) : GKCH_;

  const int ldr = tid >> 1;         // 0..127
  const int ldk = (tid & 1) * 16;   // 16 floats = 64B per thread per tensor

  const unsigned lda0 = lds_off(&As[0 * GBUF_ + ldr * GROW_ + ldk]);
  const unsigned lda1 = lds_off(&As[1 * GBUF_ + ldr * GROW_ + ldk]);
  const unsigned ldb0 = lds_off(&Bs[0 * GBUF_ + ldr * GROW_ + ldk]);
  const unsigned ldb1 = lds_off(&Bs[1 * GBUF_ + ldr * GROW_ + ldk]);
  const unsigned long long gA =
      (unsigned long long)(M + (size_t)(c0 + ldr) * HWP_ + ldk);
  const unsigned long long gB =
      (unsigned long long)(M + (size_t)(d0 + ldr) * HWP_ + ldk);

  v8f acc[2][4];
  const v8f vzero = {0.f, 0.f, 0.f, 0.f, 0.f, 0.f, 0.f, 0.f};
#pragma unroll
  for (int tm = 0; tm < 2; ++tm)
#pragma unroll
    for (int tn = 0; tn < 4; ++tn) acc[tm][tn] = vzero;

  // prologue: stage chunk kbeg into buffer 0
  async_ld64(lda0, gA + (unsigned long long)kbeg * GKC_ * 4);
  async_ld64(ldb0, gB + (unsigned long long)kbeg * GKC_ * 4);

  int cur = 0;
  for (int kc = kbeg; kc < kend; ++kc) {
    wait_async0();
    __syncthreads();
    if (kc + 1 < kend) {
      unsigned long long off = (unsigned long long)(kc + 1) * GKC_ * 4;
      async_ld64(cur ? lda0 : lda1, gA + off);
      async_ld64(cur ? ldb0 : ldb1, gB + off);
    }
#pragma unroll
    for (int ks = 0; ks < GKC_ / 4; ++ks) {
      const int kb = ks * 4 + hi * 2;  // f32 A 16x4: lanes<16 K=0,1; lanes>=16 K=2,3
      v2f a2[2], b2[4];
#pragma unroll
      for (int tm = 0; tm < 2; ++tm)
        a2[tm] = *reinterpret_cast<const v2f*>(
            &As[cur * GBUF_ + (wm * 32 + tm * 16 + lm) * GROW_ + kb]);
#pragma unroll
      for (int tn = 0; tn < 4; ++tn)
        b2[tn] = *reinterpret_cast<const v2f*>(
            &Bs[cur * GBUF_ + (wn * 64 + tn * 16 + lm) * GROW_ + kb]);
#pragma unroll
      for (int tm = 0; tm < 2; ++tm)
#pragma unroll
        for (int tn = 0; tn < 4; ++tn)
          acc[tm][tn] = __builtin_amdgcn_wmma_f32_16x16x4_f32(
              false, a2[tm], false, b2[tn], (short)0, acc[tm][tn], false, false);
    }
    cur ^= 1;
  }

  float* dst = part + (size_t)kseg * (C_ * C_);
#pragma unroll
  for (int tm = 0; tm < 2; ++tm)
#pragma unroll
    for (int tn = 0; tn < 4; ++tn)
#pragma unroll
      for (int r = 0; r < 8; ++r) {
        int row = c0 + wm * 32 + tm * 16 + hi * 8 + r;
        int col = d0 + wn * 64 + tn * 16 + lm;
        dst[(size_t)row * C_ + col] = acc[tm][tn][r];
      }
}

// ============ kernel 7: reduce split-K partials, scale by 1/(C*sum(mask)) ============
__global__ __launch_bounds__(256) void gram_reduce_k(
    const float* __restrict__ part, const float* __restrict__ msum,
    float* __restrict__ out) {
  int i = blockIdx.x * blockDim.x + threadIdx.x;
  if (i >= C_ * C_) return;
  float s = 0.f;
#pragma unroll
  for (int k = 0; k < NKSEG_; ++k) s += part[(size_t)k * (C_ * C_) + i];
  out[i] = s / (512.0f * msum[0]);
}

// ============ host launcher ============
extern "C" void kernel_launch(void* const* d_in, const int* in_sizes, int n_in,
                              void* d_out, int out_size, void* d_ws, size_t ws_size,
                              hipStream_t stream) {
  const float* style = (const float*)d_in[0];
  const float* img   = (const float*)d_in[1];
  const float* mask  = (const float*)d_in[2];
  float* out = (float*)d_out;
  char* ws = (char*)d_ws;

  unsigned short* styb = (unsigned short*)(ws);
  unsigned short* imgb = (unsigned short*)(ws + OFF_IMG);
  float* M    = (float*)(ws);              // aliases bf16 bufs (dead after argmax)
  int*   idx  = (int*)(ws + OFF_IDX);
  float* msum = (float*)(ws + OFF_MSUM);
  float* cv   = (float*)(ws + OFF_CV);
  int*   ci   = (int*)(ws + OFF_CI);
  float* part = (float*)(ws + OFF_PART);

  pack_patches_k<<<8192, 256, 0, stream>>>(style, img, styb, imgb);
  scores_argmax_k<<<dim3(56, NSEG_), 256, 0, stream>>>(imgb, styb, cv, ci);
  merge_argmax_k<<<(P_ + 255) / 256, 256, 0, stream>>>(cv, ci, idx);
  gather_mask_k<<<8192, 256, 0, stream>>>(style, mask, idx, M);
  mask_sum_k<<<1, 256, 0, stream>>>(mask, msum);
  gram_partial_k<<<dim3(16, NKSEG_), 256, 0, stream>>>(M, part);
  gram_reduce_k<<<(C_ * C_ + 255) / 256, 256, 0, stream>>>(part, msum, out);
}
